// LevelFusion_52948356825740
// MI455X (gfx1250) — matrix operand order repaired and checked by reading, not compiled, not run on hardware
//
#include <hip/hip_runtime.h>
#include <cstdint>
#include <cstddef>

typedef __bf16 bf16_t;
typedef __attribute__((ext_vector_type(16))) __bf16 v16bf;
typedef __attribute__((ext_vector_type(8)))  float  v8f;
typedef __attribute__((ext_vector_type(4)))  unsigned int v4u;
typedef __attribute__((ext_vector_type(8)))  int v8i_t;
typedef __attribute__((ext_vector_type(4)))  int v4i_t;

#define NNODE 20000
#define NEDGE 320000
#define KHALF 10000   // NNODE/2

// CDNA5 Tensor Data Mover path (TENSORcnt-tracked DMA, value-typed builtin --
// no pointer address-space issues). Guarded; falls back to sync VGPR copies.
// NOTE: this toolchain exposes the 6-arg form:
//   (uint32x4 g0, int32x8 g1, int32x4 g2, int32x4 g3, int32x8 g4, i32 cpol)
#if defined(__gfx1250__) && __has_builtin(__builtin_amdgcn_tensor_load_to_lds) && \
    __has_builtin(__builtin_amdgcn_s_wait_tensorcnt)
#define USE_TDM 1
#else
#define USE_TDM 0
#endif

// ---------- helpers ----------
static __device__ __forceinline__ unsigned orderkey(float f) {
  unsigned u = __float_as_uint(f);
  return (u & 0x80000000u) ? ~u : (u | 0x80000000u);
}
static __device__ __forceinline__ float keyfloat(unsigned u) {
  return (u & 0x80000000u) ? __uint_as_float(u & 0x7fffffffu) : __uint_as_float(~u);
}
static __device__ __forceinline__ float leaky02(float x) { return x > 0.f ? x : 0.2f * x; }
static __device__ __forceinline__ float sigmoidf(float x) { return 1.f / (1.f + __expf(-x)); }

// ---------- WMMA 16x16 tile over a single contiguous A block ----------
template <int K>
static __device__ __forceinline__ void wmma_tile_bf16(
    const bf16_t* __restrict__ ldsA,
    const bf16_t* __restrict__ Wt, int cb,
    int lane, v8f& acc)
{
  const int row = lane & 15;
  const int hi  = lane >> 4;
  const bf16_t* Arow = ldsA + (size_t)row * K;
  const bf16_t* Bcol = Wt + (size_t)(cb + row) * K + hi * 16;
#pragma unroll
  for (int k0 = 0; k0 < K; k0 += 32) {
    union { v16bf v; bf16_t e[16]; } a, b;
    const bf16_t* A0 = Arow + k0 + hi * 16;
#pragma unroll
    for (int j = 0; j < 16; ++j) a.e[j] = A0[j];
    const bf16_t* B0 = Bcol + k0;
#pragma unroll
    for (int t = 0; t < 16; ++t) b.e[t] = B0[t];
    acc = __builtin_amdgcn_wmma_f32_16x16x32_bf16(false, a.v, false, b.v,
                                                  (short)0, acc, false, false);
  }
}

// ---------- WMMA 16x16 tile over SEGMENTED A blocks (src|dst|edge) ----------
// Block selection is compile-time folded by the unrolled K loop.
template <int XC, int EC>
static __device__ __forceinline__ void wmma_tile_seg(
    const bf16_t* __restrict__ S, const bf16_t* __restrict__ D,
    const bf16_t* __restrict__ Eb,
    const bf16_t* __restrict__ Wt, int cb,
    int lane, v8f& acc)
{
  constexpr int K = 2 * XC + EC;
  const int row = lane & 15;
  const int hi  = lane >> 4;
  const bf16_t* Bcol = Wt + (size_t)(cb + row) * K + hi * 16;
#pragma unroll
  for (int k0 = 0; k0 < K; k0 += 32) {
    const bf16_t* base;
    int stride, kk;
    if (k0 < XC)            { base = S;  stride = XC; kk = k0; }
    else if (k0 < 2 * XC)   { base = D;  stride = XC; kk = k0 - XC; }
    else                    { base = Eb; stride = EC; kk = k0 - 2 * XC; }
    union { v16bf v; bf16_t e[16]; } a, b;
    const bf16_t* A0 = base + row * stride + kk + hi * 16;
#pragma unroll
    for (int j = 0; j < 16; ++j) a.e[j] = A0[j];
    const bf16_t* B0 = Bcol + k0;
#pragma unroll
    for (int t = 0; t < 16; ++t) b.e[t] = B0[t];
    acc = __builtin_amdgcn_wmma_f32_16x16x32_bf16(false, a.v, false, b.v,
                                                  (short)0, acc, false, false);
  }
}

// ---------- generic edge-conv: m = [x_src | x_dst | e] @ W + b ----------
template <int XC, int EC, int NOUT>
__global__ __launch_bounds__(256) void edge_conv_wmma(
    const bf16_t* __restrict__ xin,                          // node input, row stride 256
    const bf16_t* __restrict__ ein, int e_stride, int e_off,
    const bf16_t* __restrict__ Wt,                           // N-major bf16 weights
    const float* __restrict__ bias,
    const int* __restrict__ srcI, const int* __restrict__ dstI, int nE,
    float* __restrict__ nodeAcc,
    bf16_t* __restrict__ mOut1, int m1_stride, int m1_off,
    bf16_t* __restrict__ mOut2, int m2_stride, int m2_off,
    const float* __restrict__ emask)
{
  constexpr int K = 2 * XC + EC;
  __shared__ bf16_t ldsS[16 * XC];
  __shared__ bf16_t ldsD[16 * XC];
  __shared__ bf16_t ldsE[16 * EC];
  const int tid = threadIdx.x;
  const int lane = tid & 31, wave = tid >> 5;
  const int e0 = blockIdx.x * 16;

  // hint next tile's index cachelines into cache (global_prefetch_b8)
  if (tid < 2) {
    int ne = min(e0 + 16, nE - 1);
    __builtin_prefetch(tid ? (const void*)&dstI[ne] : (const void*)&srcI[ne], 0, 1);
  }

#if USE_TDM
  // ---- edge-feature tile: one Tensor-DMA 2D tile load (wave 0 only) ----
  if (tid < 32) {
    unsigned long long ga =
        (unsigned long long)(uintptr_t)(ein + (size_t)e0 * e_stride + e_off);
    unsigned ldsoff = (unsigned)(uintptr_t)(&ldsE[0]);   // LDS byte offset
    unsigned rows = (unsigned)(nE - e0);                 // OOB rows read as 0
    v4u g0;
    g0.x = 1u;                                           // count=1, gather off
    g0.y = ldsoff;                                       // lds_addr
    g0.z = (unsigned)(ga & 0xffffffffu);                 // global_addr[31:0]
    g0.w = (unsigned)(((ga >> 32) & 0x1ffffffu) | (2u << 30)); // addr hi | type=2
    v8i_t g1;
    g1[0] = (int)(1u << 16);                             // data_size=2B, mask=0
    g1[1] = (int)(((unsigned)EC & 0xffffu) << 16);       // tensor_dim0[15:0]=EC
    g1[2] = (int)((rows & 0xffffu) << 16);               // tensor_dim1[15:0]
    g1[3] = (int)(((rows >> 16) & 0xffffu) |
                  ((unsigned)EC << 16));                 // dim1 hi | tile_dim0=EC
    g1[4] = 16;                                          // tile_dim1=16 rows
    g1[5] = (int)e_stride;                               // dim0_stride[31:0]
    g1[6] = 0;
    g1[7] = 0;
    v4i_t g2 = {0, 0, 0, 0}, g3 = {0, 0, 0, 0};
    v8i_t g4 = {0, 0, 0, 0, 0, 0, 0, 0};
    __builtin_amdgcn_tensor_load_to_lds(g0, g1, g2, g3, g4, 0);
    __builtin_amdgcn_s_wait_tensorcnt(0);
  }
#endif

  // ---- cooperative gather of src/dst rows into LDS (16B chunks) ----
  constexpr int CPRX = XC >> 3;
#pragma unroll
  for (int c = tid; c < 16 * CPRX; c += 256) {
    int er = c / CPRX, ch = (c - er * CPRX) << 3;
    int edge = min(e0 + er, nE - 1);
    *(uint4*)(ldsS + er * XC + ch) = *(const uint4*)(xin + (size_t)srcI[edge] * 256 + ch);
  }
#pragma unroll
  for (int c = tid; c < 16 * CPRX; c += 256) {
    int er = c / CPRX, ch = (c - er * CPRX) << 3;
    int edge = min(e0 + er, nE - 1);
    *(uint4*)(ldsD + er * XC + ch) = *(const uint4*)(xin + (size_t)dstI[edge] * 256 + ch);
  }
#if !USE_TDM
  constexpr int CPRE = EC >> 3;
#pragma unroll
  for (int c = tid; c < 16 * CPRE; c += 256) {
    int er = c / CPRE, ch = (c - er * CPRE) << 3;
    int edge = min(e0 + er, nE - 1);
    *(uint4*)(ldsE + er * EC + ch) =
        *(const uint4*)(ein + (size_t)edge * e_stride + e_off + ch);
  }
#endif
  __syncthreads();

  // ---- 8 waves cover NOUT in 16-wide column tiles ----
#pragma unroll
  for (int nt = wave; nt < (NOUT >> 4); nt += 8) {
    const int cb = nt << 4;
    v8f acc = {};
    wmma_tile_seg<XC, EC>(ldsS, ldsD, ldsE, Wt, cb, lane, acc);

    const int hi = lane >> 4, col = lane & 15;
    const int ocol = cb + col;
    const float bv = bias[ocol];
#pragma unroll
    for (int r = 0; r < 8; ++r) {
      const int edge = e0 + r + hi * 8;
      if (edge < nE) {
        float mval = acc[r] + bv;
        float scale = emask ? emask[edge] : 1.0f;
        atomicAdd(&nodeAcc[(size_t)dstI[edge] * NOUT + ocol], mval * scale);
        if (mOut1) {
          bf16_t rv = (bf16_t)fmaxf(mval, 0.f);
          mOut1[(size_t)edge * m1_stride + m1_off + ocol] = rv;
          if (mOut2) mOut2[(size_t)edge * m2_stride + m2_off + ocol] = rv;
        }
      }
    }
  }
}

// ---------- node GEMM: out_bf16 = xin @ W + b (no ReLU) ----------
template <int K, int NOUT>
__global__ __launch_bounds__(256) void node_gemm_wmma(
    const bf16_t* __restrict__ xin, int in_stride,
    const bf16_t* __restrict__ Wt, const float* __restrict__ bias,
    int nRows,
    bf16_t* __restrict__ out, int out_stride, int out_off)
{
  __shared__ bf16_t ldsA[16 * K];
  const int tid = threadIdx.x;
  const int lane = tid & 31, wave = tid >> 5;
  const int r0 = blockIdx.x * 16;

  constexpr int chunksPerRow = K >> 3;
#pragma unroll
  for (int c = tid; c < 16 * chunksPerRow; c += 256) {
    int er = c / chunksPerRow, ch = (c - er * chunksPerRow) << 3;
    int row = min(r0 + er, nRows - 1);
    *(uint4*)(ldsA + (size_t)er * K + ch) =
        *(const uint4*)(xin + (size_t)row * in_stride + ch);
  }
  __syncthreads();

#pragma unroll
  for (int nt = wave; nt < (NOUT >> 4); nt += 8) {
    const int cb = nt << 4;
    v8f acc = {};
    wmma_tile_bf16<K>(ldsA, Wt, cb, lane, acc);
    const int hi = lane >> 4, col = lane & 15;
    const int ocol = cb + col;
    const float bv = bias[ocol];
#pragma unroll
    for (int r = 0; r < 8; ++r) {
      const int row = r0 + r + hi * 8;
      if (row < nRows)
        out[(size_t)row * out_stride + out_off + ocol] = (bf16_t)(acc[r] + bv);
    }
  }
}

// ---------- elementwise / aggregation helpers ----------
__global__ void convert_bf16_k(const float* __restrict__ s, bf16_t* __restrict__ d, int n) {
  int i = blockIdx.x * blockDim.x + threadIdx.x;
  if (i < n) d[i] = (bf16_t)s[i];
}

// weights: row-major f32 (K x Nout) -> N-major bf16 (Nout x K)
__global__ void convert_w_transpose(const float* __restrict__ W, bf16_t* __restrict__ Wt,
                                    int K, int Nout) {
  int idx = blockIdx.x * blockDim.x + threadIdx.x;
  if (idx >= K * Nout) return;
  int k = idx / Nout, n = idx - k * Nout;
  Wt[(size_t)n * K + k] = (bf16_t)W[idx];
}

__global__ void deg_kernel(const int* __restrict__ dst, int nE, float* __restrict__ deg) {
  int e = blockIdx.x * blockDim.x + threadIdx.x;
  if (e < nE) atomicAdd(&deg[dst[e]], 1.0f);
}

__global__ void mean_store(const float* __restrict__ acc, const float* __restrict__ deg,
                           int n, int Nout, int doRelu,
                           bf16_t* __restrict__ o1, int s1, int f1,
                           bf16_t* __restrict__ o2, int s2, int f2,
                           float* __restrict__ of) {
  int idx = blockIdx.x * blockDim.x + threadIdx.x;
  if (idx >= n * Nout) return;
  int row = idx / Nout, c = idx - row * Nout;
  float v = acc[idx] / fmaxf(deg[row], 1.f);
  if (doRelu) v = fmaxf(v, 0.f);
  if (o1) o1[(size_t)row * s1 + f1 + c] = (bf16_t)v;
  if (o2) o2[(size_t)row * s2 + f2 + c] = (bf16_t)v;
  if (of) of[idx] = v;
}

__global__ void store_bf16_strided(const float* __restrict__ s, int n, int cols,
                                   bf16_t* __restrict__ d, int stride, int off) {
  int idx = blockIdx.x * blockDim.x + threadIdx.x;
  if (idx >= n) return;
  int row = idx / cols, c = idx - row * cols;
  d[(size_t)row * stride + off + c] = (bf16_t)s[idx];
}

__global__ void dot_h(const float* __restrict__ xs1f, const float* __restrict__ Wg,
                      int n, float* __restrict__ h) {
  int i = blockIdx.x * blockDim.x + threadIdx.x;
  if (i >= n) return;
  float s = 0.f;
  for (int c = 0; c < 256; ++c) s += xs1f[(size_t)i * 256 + c] * Wg[c];
  h[i] = s;
}

// GAT softmax passes over E edges + N self-loops
__global__ void att_max(const int* src, const int* dst, int nE, int n,
                        const float* __restrict__ h, const float* as, const float* ad,
                        unsigned* __restrict__ emaxk) {
  int i = blockIdx.x * blockDim.x + threadIdx.x;
  if (i >= nE + n) return;
  int s, d;
  if (i < nE) { s = src[i]; d = dst[i]; } else { s = d = i - nE; }
  float ea = leaky02(h[s] * as[0] + h[d] * ad[0]);
  atomicMax(&emaxk[d], orderkey(ea));
}
__global__ void att_sum(const int* src, const int* dst, int nE, int n,
                        const float* __restrict__ h, const float* as, const float* ad,
                        const unsigned* __restrict__ emaxk, float* __restrict__ denom) {
  int i = blockIdx.x * blockDim.x + threadIdx.x;
  if (i >= nE + n) return;
  int s, d;
  if (i < nE) { s = src[i]; d = dst[i]; } else { s = d = i - nE; }
  float ea = leaky02(h[s] * as[0] + h[d] * ad[0]);
  atomicAdd(&denom[d], __expf(ea - keyfloat(emaxk[d])));
}
__global__ void att_score(const int* src, const int* dst, int nE, int n,
                          const float* __restrict__ h, const float* as, const float* ad,
                          const unsigned* __restrict__ emaxk, const float* __restrict__ denom,
                          float* __restrict__ score) {
  int i = blockIdx.x * blockDim.x + threadIdx.x;
  if (i >= nE + n) return;
  int s, d;
  if (i < nE) { s = src[i]; d = dst[i]; } else { s = d = i - nE; }
  float ea = leaky02(h[s] * as[0] + h[d] * ad[0]);
  float w = __expf(ea - keyfloat(emaxk[d])) / denom[d];
  atomicAdd(&score[d], w * h[s]);
}

// device-side top-k (radix select on ordered keys, descending)
__global__ void key_kernel(const float* __restrict__ s, unsigned* __restrict__ key, int n) {
  int i = blockIdx.x * blockDim.x + threadIdx.x;
  if (i < n) key[i] = orderkey(s[i]);
}
__global__ void init_state(unsigned* state, unsigned* hist, int k) {
  if (threadIdx.x == 0 && blockIdx.x == 0) {
    state[0] = 0u; state[1] = (unsigned)k; state[3] = 0u; state[4] = 0u;
    for (int j = 0; j < 256; ++j) hist[j] = 0u;
  }
}
__global__ void hist_kernel(const unsigned* __restrict__ key, int n, int shift,
                            const unsigned* __restrict__ state, unsigned* __restrict__ hist) {
  int i = blockIdx.x * blockDim.x + threadIdx.x;
  if (i >= n) return;
  unsigned u = key[i];
  if (shift == 24 || (u >> (shift + 8)) == (state[0] >> (shift + 8)))
    atomicAdd(&hist[(u >> shift) & 255u], 1u);
}
__global__ void select_bin_kernel(unsigned* hist, unsigned* state, int shift) {
  if (threadIdx.x == 0 && blockIdx.x == 0) {
    unsigned rem = state[1], cum = 0;
    for (int bin = 255; bin >= 0; --bin) {
      unsigned hh = hist[bin];
      if (cum + hh >= rem) { state[0] |= ((unsigned)bin) << shift; state[1] = rem - cum; break; }
      cum += hh;
    }
    for (int j = 0; j < 256; ++j) hist[j] = 0u;
  }
}
__global__ void select_kernel(const unsigned* __restrict__ key, int n, unsigned* state,
                              int k, int* __restrict__ perm, int* __restrict__ nidx) {
  int i = blockIdx.x * blockDim.x + threadIdx.x;
  if (i >= n) return;
  unsigned T = state[0], need = state[1], g = (unsigned)k - need;
  unsigned u = key[i];
  if (u > T) {
    unsigned p = atomicAdd(&state[3], 1u);
    if (p < (unsigned)k) { perm[p] = i; nidx[i] = (int)p; }
  } else if (u == T) {
    unsigned t = atomicAdd(&state[4], 1u);
    if (t < need) { unsigned p = g + t; perm[p] = i; nidx[i] = (int)p; }
  }
}

__global__ void build_xp(const int* __restrict__ perm, const float* __restrict__ score,
                         const float* __restrict__ xs1f, int k, bf16_t* __restrict__ out) {
  int idx = blockIdx.x * blockDim.x + threadIdx.x;
  if (idx >= k * 256) return;
  int i = idx >> 8, c = idx & 255;
  int node = perm[i];
  float g = tanhf(score[node]);
  out[(size_t)i * 256 + c] = (bf16_t)(xs1f[(size_t)node * 256 + c] * g);
}

__global__ void remap_edges(const int* __restrict__ src, const int* __restrict__ dst, int nE,
                            const int* __restrict__ nidx, int* __restrict__ sp,
                            int* __restrict__ dp, float* __restrict__ emk,
                            float* __restrict__ degm) {
  int e = blockIdx.x * blockDim.x + threadIdx.x;
  if (e >= nE) return;
  int ns = nidx[src[e]], nd = nidx[dst[e]];
  int keep = (ns >= 0) && (nd >= 0);
  sp[e] = keep ? ns : 0;
  dp[e] = keep ? nd : 0;
  float m = keep ? 1.f : 0.f;
  emk[e] = m;
  atomicAdd(&degm[dp[e]], m);
}

__global__ void scatter_add(const int* __restrict__ perm, const float* __restrict__ xs2f,
                            int k, float* __restrict__ x32f) {
  int idx = blockIdx.x * blockDim.x + threadIdx.x;
  if (idx >= k * 128) return;
  int i = idx >> 7, c = idx & 127;
  x32f[(size_t)perm[i] * 128 + c] += xs2f[idx];
}

__global__ void out_x_kernel(const float* __restrict__ x4f, const float* __restrict__ Wl1,
                             const float* __restrict__ bl1, int n, float* __restrict__ out) {
  int i = blockIdx.x * blockDim.x + threadIdx.x;
  if (i >= n) return;
  float s = bl1[0];
  for (int c = 0; c < 128; ++c) s += x4f[(size_t)i * 128 + c] * Wl1[c];
  out[i] = sigmoidf(s);
}
__global__ void out_w_kernel(const bf16_t* __restrict__ e4, const float* __restrict__ Ww,
                             const float* __restrict__ bw, int nE, float* __restrict__ out) {
  int e = blockIdx.x * blockDim.x + threadIdx.x;
  if (e >= nE) return;
  float s = bw[0];
  for (int c = 0; c < 128; ++c) s += (float)e4[(size_t)e * 256 + c] * Ww[c];
  out[e] = sigmoidf(s);
}

// =====================================================================
extern "C" void kernel_launch(void* const* d_in, const int* in_sizes, int n_in,
                              void* d_out, int out_size, void* d_ws, size_t ws_size,
                              hipStream_t stream) {
  (void)in_sizes; (void)n_in; (void)out_size; (void)ws_size;
  const int N = NNODE, E = NEDGE, K2 = KHALF;

  const float* node_feat = (const float*)d_in[0];
  const float* edge_feat = (const float*)d_in[1];
  const int*   ei        = (const int*)d_in[2];
  const int *src = ei, *dst = ei + E;
  const float *W_red = (const float*)d_in[3],  *b_red = (const float*)d_in[4];
  const float *W1 = (const float*)d_in[5],  *b1 = (const float*)d_in[6];
  const float *W2 = (const float*)d_in[7],  *b2 = (const float*)d_in[8];
  const float *W3 = (const float*)d_in[9],  *b3 = (const float*)d_in[10];
  const float *Wpre = (const float*)d_in[11], *bpre = (const float*)d_in[12];
  const float *Wsub = (const float*)d_in[13], *bsub = (const float*)d_in[14];
  const float *W4 = (const float*)d_in[15], *b4 = (const float*)d_in[16];
  const float *Wg = (const float*)d_in[17];
  const float *a_s = (const float*)d_in[18], *a_d = (const float*)d_in[19];
  const float *Wl1 = (const float*)d_in[20], *bl1 = (const float*)d_in[21];
  const float *Ww = (const float*)d_in[22],  *bw = (const float*)d_in[23];
  float* out = (float*)d_out;

  // ---- workspace layout ----
  char* wsp = (char*)d_ws;
  size_t off = 0;
  auto alloc = [&](size_t bytes) -> char* {
    char* p = wsp + off;
    off = (off + bytes + 255) & ~(size_t)255;
    return p;
  };
  bf16_t* nfbuf = (bf16_t*)alloc((size_t)N * 256 * 2);
  bf16_t* xinA  = (bf16_t*)alloc((size_t)N * 256 * 2);
  bf16_t* xinB  = (bf16_t*)alloc((size_t)N * 256 * 2);
  bf16_t* ebuf0 = (bf16_t*)alloc((size_t)E * 128 * 2);
  bf16_t* ecat  = (bf16_t*)alloc((size_t)E * 256 * 2);
  bf16_t* ecat2 = (bf16_t*)alloc((size_t)E * 256 * 2);
  bf16_t* es1b  = (bf16_t*)alloc((size_t)E * 256 * 2);
  bf16_t* Wredb = (bf16_t*)alloc((size_t)256 * 128 * 2);
  bf16_t* W1b   = (bf16_t*)alloc((size_t)384 * 128 * 2);
  bf16_t* W2b   = (bf16_t*)alloc((size_t)384 * 128 * 2);
  bf16_t* W3b   = (bf16_t*)alloc((size_t)768 * 128 * 2);
  bf16_t* Wpreb = (bf16_t*)alloc((size_t)384 * 256 * 2);
  bf16_t* Wsubb = (bf16_t*)alloc((size_t)768 * 128 * 2);
  bf16_t* W4b   = (bf16_t*)alloc((size_t)640 * 128 * 2);
  float* acc  = (float*)alloc((size_t)N * 256 * 4);
  float* deg  = (float*)alloc((size_t)N * 4);
  float* degm = (float*)alloc((size_t)N * 4);
  float* x3f  = (float*)alloc((size_t)N * 128 * 4);
  float* x32f = (float*)alloc((size_t)N * 128 * 4);
  float* xs1f = (float*)alloc((size_t)N * 256 * 4);
  float* xs2f = (float*)alloc((size_t)K2 * 128 * 4);
  float* x4f  = (float*)alloc((size_t)N * 128 * 4);
  float* hbuf = (float*)alloc((size_t)N * 4);
  float* denom = (float*)alloc((size_t)N * 4);
  float* score = (float*)alloc((size_t)N * 4);
  unsigned* emaxk = (unsigned*)alloc((size_t)N * 4);
  unsigned* keys  = (unsigned*)alloc((size_t)N * 4);
  int* perm = (int*)alloc((size_t)K2 * 4);
  int* nidx = (int*)alloc((size_t)N * 4);
  int* sp   = (int*)alloc((size_t)E * 4);
  int* dp   = (int*)alloc((size_t)E * 4);
  float* emk = (float*)alloc((size_t)E * 4);
  unsigned* hist  = (unsigned*)alloc(256 * 4);
  unsigned* state = (unsigned*)alloc(8 * 4);

  auto cvt = [&](const float* s, bf16_t* d, size_t n) {
    convert_bf16_k<<<dim3((unsigned)((n + 255) / 256)), 256, 0, stream>>>(s, d, (int)n);
  };
  auto cvtW = [&](const float* W, bf16_t* Wt, int Kd, int Nd) {
    convert_w_transpose<<<dim3((Kd * Nd + 255) / 256), 256, 0, stream>>>(W, Wt, Kd, Nd);
  };
  // bf16 conversions (inputs) + N-major bf16 weights
  cvt(node_feat, nfbuf, (size_t)N * 256);
  cvt(edge_feat, ebuf0, (size_t)E * 128);
  cvtW(W_red, Wredb, 256, 128); cvtW(W1, W1b, 384, 128); cvtW(W2, W2b, 384, 128);
  cvtW(W3, W3b, 768, 128);      cvtW(Wpre, Wpreb, 384, 256);
  cvtW(Wsub, Wsubb, 768, 128);  cvtW(W4, W4b, 640, 128);

  const dim3 egrid((E + 15) / 16);
  auto zeroAcc = [&](size_t elems) { (void)hipMemsetAsync(acc, 0, elems * 4, stream); };
  auto meanN = [&](int n, int Nout, bf16_t* o1, int f1, bf16_t* o2, int f2,
                   float* of, const float* dg) {
    mean_store<<<dim3((n * Nout + 255) / 256), 256, 0, stream>>>(
        acc, dg, n, Nout, 1, o1, 256, f1, o2, 256, f2, of);
  };

  // degrees (shared by all unmasked convs)
  (void)hipMemsetAsync(deg, 0, (size_t)N * 4, stream);
  deg_kernel<<<dim3((E + 255) / 256), 256, 0, stream>>>(dst, E, deg);

  // x0 = node_feat @ W_red + b  -> xinA[:,0:128]
  node_gemm_wmma<256, 128><<<dim3((N + 15) / 16), 256, 0, stream>>>(
      nfbuf, 256, Wredb, b_red, N, xinA, 256, 0);

  // conv1: [x0|x0|edge_feat](384) -> e1 -> ecat[:,128:256]
  zeroAcc((size_t)N * 128);
  edge_conv_wmma<128, 128, 128><<<egrid, 256, 0, stream>>>(
      xinA, ebuf0, 128, 0, W1b, b1, src, dst, E, acc,
      ecat, 256, 128, nullptr, 0, 0, nullptr);
  meanN(N, 128, xinB, 0, xinA, 128, nullptr, deg);        // x1 -> xinB[:,:128], xinA[:,128:]

  // conv2: e-in = e1 (ecat off128) -> e2 -> ecat[:,0:128] and ecat2[:,128:256]
  zeroAcc((size_t)N * 128);
  edge_conv_wmma<128, 128, 128><<<egrid, 256, 0, stream>>>(
      xinB, ecat, 256, 128, W2b, b2, src, dst, E, acc,
      ecat, 256, 0, ecat2, 256, 128, nullptr);
  meanN(N, 128, xinA, 0, xinB, 128, nullptr, deg);        // x2 -> xinA[:,:128], xinB[:,128:]

  // conv3a: x-in=[x2|x1] e-in=[e2|e1] -> e3a -> ecat2[:,0:128]
  zeroAcc((size_t)N * 128);
  edge_conv_wmma<256, 256, 128><<<egrid, 256, 0, stream>>>(
      xinA, ecat, 256, 0, W3b, b3, src, dst, E, acc,
      ecat2, 256, 0, nullptr, 0, 0, nullptr);
  meanN(N, 128, xinB, 0, nullptr, 0, nullptr, deg);       // x3a -> xinB[:,:128]

  // conv3b (shared W3): x-in=[x3a|x2] e-in=[e3a|e2] -> e3 -> ebuf0
  zeroAcc((size_t)N * 128);
  edge_conv_wmma<256, 256, 128><<<egrid, 256, 0, stream>>>(
      xinB, ecat2, 256, 0, W3b, b3, src, dst, E, acc,
      ebuf0, 128, 0, nullptr, 0, 0, nullptr);
  meanN(N, 128, xinA, 0, xinA, 128, x3f, deg);            // x3 -> xinA both halves + f32

  // pre conv: x-in=x3, e-in=e3, Nout=256 -> es1
  zeroAcc((size_t)N * 256);
  edge_conv_wmma<128, 128, 256><<<egrid, 256, 0, stream>>>(
      xinA, ebuf0, 128, 0, Wpreb, bpre, src, dst, E, acc,
      es1b, 256, 0, nullptr, 0, 0, nullptr);
  meanN(N, 256, nullptr, 0, nullptr, 0, xs1f, deg);       // xs1 f32

  // ---- SAGPool scoring (GAT softmax with self-loops) ----
  dot_h<<<dim3((N + 255) / 256), 256, 0, stream>>>(xs1f, Wg, N, hbuf);
  (void)hipMemsetAsync(emaxk, 0, (size_t)N * 4, stream);
  (void)hipMemsetAsync(denom, 0, (size_t)N * 4, stream);
  (void)hipMemsetAsync(score, 0, (size_t)N * 4, stream);
  const dim3 agrid((E + N + 255) / 256);
  att_max<<<agrid, 256, 0, stream>>>(src, dst, E, N, hbuf, a_s, a_d, emaxk);
  att_sum<<<agrid, 256, 0, stream>>>(src, dst, E, N, hbuf, a_s, a_d, emaxk, denom);
  att_score<<<agrid, 256, 0, stream>>>(src, dst, E, N, hbuf, a_s, a_d, emaxk, denom, score);

  // ---- device-side top-k (radix select) ----
  key_kernel<<<dim3((N + 255) / 256), 256, 0, stream>>>(score, keys, N);
  init_state<<<1, 1, 0, stream>>>(state, hist, K2);
  for (int shift = 24; shift >= 0; shift -= 8) {
    hist_kernel<<<dim3((N + 255) / 256), 256, 0, stream>>>(keys, N, shift, state, hist);
    select_bin_kernel<<<1, 1, 0, stream>>>(hist, state, shift);
  }
  (void)hipMemsetAsync(nidx, 0xFF, (size_t)N * 4, stream);
  select_kernel<<<dim3((N + 255) / 256), 256, 0, stream>>>(keys, N, state, K2, perm, nidx);

  // xp = xs1[perm] * tanh(topv)  -> xinB rows [0,k)
  build_xp<<<dim3((K2 * 256 + 255) / 256), 256, 0, stream>>>(perm, score, xs1f, K2, xinB);
  (void)hipMemsetAsync(degm, 0, (size_t)N * 4, stream);
  remap_edges<<<dim3((E + 255) / 256), 256, 0, stream>>>(src, dst, E, nidx, sp, dp, emk, degm);

  // sub conv (masked): x-in=xp(256), e-in=es1(256) -> xs2
  zeroAcc((size_t)N * 256);
  edge_conv_wmma<256, 256, 128><<<egrid, 256, 0, stream>>>(
      xinB, es1b, 256, 0, Wsubb, bsub, sp, dp, E, acc,
      nullptr, 0, 0, nullptr, 0, 0, emk);
  mean_store<<<dim3((K2 * 128 + 255) / 256), 256, 0, stream>>>(
      acc, degm, K2, 128, 1, nullptr, 0, 0, nullptr, 0, 0, xs2f);

  // x3_2 = x3; x3_2[perm] += xs2  -> bf16 xinA[:,0:128]
  (void)hipMemcpyAsync(x32f, x3f, (size_t)N * 128 * 4, hipMemcpyDeviceToDevice, stream);
  scatter_add<<<dim3((K2 * 128 + 255) / 256), 256, 0, stream>>>(perm, xs2f, K2, x32f);
  store_bf16_strided<<<dim3((N * 128 + 255) / 256), 256, 0, stream>>>(
      x32f, N * 128, 128, xinA, 256, 0);

  // conv4: x-in=[x3_2|x3](256), e-in=e3(128) -> e4 -> ecat[:,0:128]
  zeroAcc((size_t)N * 128);
  edge_conv_wmma<256, 128, 128><<<egrid, 256, 0, stream>>>(
      xinA, ebuf0, 128, 0, W4b, b4, src, dst, E, acc,
      ecat, 256, 0, nullptr, 0, 0, nullptr);
  meanN(N, 128, nullptr, 0, nullptr, 0, x4f, deg);        // x4 f32

  // heads
  out_x_kernel<<<dim3((N + 255) / 256), 256, 0, stream>>>(x4f, Wl1, bl1, N, out);
  out_w_kernel<<<dim3((E + 255) / 256), 256, 0, stream>>>(ecat, Ww, bw, E, out + N);
}
